// ResidualMLP_51513837749004
// MI455X (gfx1250) — matrix operand order repaired and checked by reading, not compiled, not run on hardware
//
#include <hip/hip_runtime.h>

typedef __attribute__((ext_vector_type(8)))  float  v8f;
typedef __attribute__((ext_vector_type(4)))  float  v4f;
typedef __attribute__((ext_vector_type(4)))  __bf16 v4bf;
typedef __attribute__((ext_vector_type(8)))  __bf16 v8bf;
typedef __attribute__((ext_vector_type(16))) __bf16 v16bf;

#define HID 512
#define DEPTH 8
#define MT 32           // rows per workgroup
#define XPAD 8          // bf16 pad -> 16B
#define XSTR (HID + XPAD)
#define HPAD 4          // f32 pad -> 16B
#define HSTR (HID + HPAD)

static __device__ __forceinline__ __bf16 f2bf(float f) {
  union { float f; unsigned int u; } v; v.f = f;
  unsigned int r = v.u + 0x7FFFu + ((v.u >> 16) & 1u);   // round-to-nearest-even
  unsigned short h = (unsigned short)(r >> 16);
  return __builtin_bit_cast(__bf16, h);
}

// Convert + transpose weights: Wt[n*512 + k] = bf16(W[k*512 + n])
__global__ __launch_bounds__(256) void prep_weights(
    const float* __restrict__ Win, const float* __restrict__ Whead,
    __bf16* __restrict__ WinT, __bf16* __restrict__ WheadT) {
  int i = blockIdx.x * 256 + threadIdx.x;      // 0 .. 512*512-1
  int k = i >> 9;
  int n = i & (HID - 1);
  WinT  [n * HID + k] = f2bf(Win  [k * HID + n]);
  WheadT[n * HID + k] = f2bf(Whead[k * HID + n]);
}

// 32x(512)x(512) bf16 GEMM: A tile in LDS (row-major, stride XSTR),
// B = Wt (N-major, K-contiguous). Each wave owns cols [w*64, w*64+64)
// for both 16-row M tiles; B fragments are reused across the two M tiles.
static __device__ __forceinline__ void gemm_tile(
    const __bf16* __restrict__ xtile,   // LDS, MT x XSTR
    const __bf16* __restrict__ Wt,      // global, bf16 [N][K]
    v8f acc[2][4], int w, int lane) {
  const int m    = lane & 15;
  const int half = lane >> 4;
  #pragma unroll 2
  for (int kk = 0; kk < HID; kk += 32) {
    // A fragments: lane<16 -> K={kk..kk+7, kk+16..kk+23}; lane>=16 -> +8
    v16bf A[2];
    #pragma unroll
    for (int mt = 0; mt < 2; ++mt) {
      const __bf16* arow = xtile + (mt * 16 + m) * XSTR + kk + half * 8;
      v8bf a0 = *(const v8bf*)(arow);
      v8bf a1 = *(const v8bf*)(arow + 16);
      A[mt] = __builtin_shufflevector(a0, a1, 0,1,2,3,4,5,6,7,8,9,10,11,12,13,14,15);
    }
    #pragma unroll
    for (int nt = 0; nt < 4; ++nt) {
      // B fragment: lane<16 -> K={kk..kk+7, kk+16..kk+23}; lane>=16 -> +8
      const int n = (w << 6) + (nt << 4) + m;
      const __bf16* brow = Wt + n * HID + kk + half * 8;
      v8bf b0 = *(const v8bf*)(brow);
      v8bf b1 = *(const v8bf*)(brow + 16);
      v16bf B = __builtin_shufflevector(b0, b1, 0,1,2,3,4,5,6,7,8,9,10,11,12,13,14,15);
      #pragma unroll
      for (int mt = 0; mt < 2; ++mt) {
        acc[mt][nt] = __builtin_amdgcn_wmma_f32_16x16x32_bf16(
            false, A[mt], false, B, (short)0, acc[mt][nt], false, false);
      }
    }
  }
}

__global__ __launch_bounds__(256) void fused_residual_mlp(
    const float* __restrict__ x,
    const __bf16* __restrict__ WinT,  const float* __restrict__ b_in,
    const float* __restrict__ ln_g,   const float* __restrict__ ln_b,
    const float* __restrict__ W1,     const float* __restrict__ b1,
    const float* __restrict__ W2,     const float* __restrict__ b2,
    const __bf16* __restrict__ WheadT,const float* __restrict__ b_head,
    float* __restrict__ out) {
  __shared__ __bf16 xbuf[MT * XSTR];   // x tile (phase 1) then h-bf16 (phase 3)
  __shared__ float  hs  [MT * HSTR];   // h f32 between phase 1 and phase 2

  const int tid  = threadIdx.x;
  const int lane = tid & 31;
  const int w    = tid >> 5;
  const long row0 = (long)blockIdx.x * MT;

  // ---- Phase 0: stage x tile into LDS as bf16 (b128 NT loads, b64 LDS stores) ----
  {
    const v4f* xv = (const v4f*)(x + row0 * HID);   // MT*128 float4s
    #pragma unroll
    for (int i = tid; i < MT * (HID / 4); i += 256) {
      const int r  = i >> 7;             // row
      const int c4 = i & 127;            // float4 index within row
      const v4f v = __builtin_nontemporal_load(&xv[r * (HID / 4) + c4]);
      v4bf b; b[0] = f2bf(v[0]); b[1] = f2bf(v[1]); b[2] = f2bf(v[2]); b[3] = f2bf(v[3]);
      *(v4bf*)&xbuf[r * XSTR + (c4 << 2)] = b;
    }
  }
  __syncthreads();

  // ---- Phase 1: h = relu(x @ W_in + b_in) ----
  v8f acc[2][4];
  #pragma unroll
  for (int mt = 0; mt < 2; ++mt)
    #pragma unroll
    for (int nt = 0; nt < 4; ++nt)
      acc[mt][nt] = v8f{0.f,0.f,0.f,0.f,0.f,0.f,0.f,0.f};
  gemm_tile(xbuf, WinT, acc, w, lane);
  #pragma unroll
  for (int mt = 0; mt < 2; ++mt) {
    #pragma unroll
    for (int nt = 0; nt < 4; ++nt) {
      const int col = (w << 6) + (nt << 4) + (lane & 15);
      const float bc = b_in[col];
      #pragma unroll
      for (int r = 0; r < 8; ++r) {
        const int row = mt * 16 + r + ((lane >> 4) << 3);
        hs[row * HSTR + col] = fmaxf(acc[mt][nt][r] + bc, 0.0f);
      }
    }
  }
  __syncthreads();

  // ---- Phase 2: 8 residual blocks; each wave owns 2 full rows per pass ----
  const int kbase = (lane & 15) << 5;            // 32 elements per lane
  for (int pass = 0; pass < 2; ++pass) {
    const int myrow = pass * 16 + (w << 1) + (lane >> 4);
    float hv[32];
    #pragma unroll
    for (int i = 0; i < 32; ++i) hv[i] = hs[myrow * HSTR + kbase + i];

    for (int d = 0; d < DEPTH; ++d) {
      // mean
      float s = 0.f;
      #pragma unroll
      for (int i = 0; i < 32; ++i) s += hv[i];
      s += __shfl_xor(s, 1); s += __shfl_xor(s, 2);
      s += __shfl_xor(s, 4); s += __shfl_xor(s, 8);
      const float mu = s * (1.0f / 512.0f);
      // variance
      float vs = 0.f;
      #pragma unroll
      for (int i = 0; i < 32; ++i) { const float t = hv[i] - mu; vs += t * t; }
      vs += __shfl_xor(vs, 1); vs += __shfl_xor(vs, 2);
      vs += __shfl_xor(vs, 4); vs += __shfl_xor(vs, 8);
      const float rstd = __frsqrt_rn(vs * (1.0f / 512.0f) + 1e-5f);

      // z = relu(layernorm(h) @ W1 + b1)   (EXP = 4)
      const float* gp  = ln_g + d * HID + kbase;
      const float* bp  = ln_b + d * HID + kbase;
      const float* w1p = W1 + ((d * HID + kbase) << 2);
      float p0 = 0.f, p1 = 0.f, p2 = 0.f, p3 = 0.f;
      #pragma unroll
      for (int i = 0; i < 32; ++i) {
        const float zn = (hv[i] - mu) * rstd * gp[i] + bp[i];
        p0 += zn * w1p[4 * i + 0];
        p1 += zn * w1p[4 * i + 1];
        p2 += zn * w1p[4 * i + 2];
        p3 += zn * w1p[4 * i + 3];
      }
      p0 += __shfl_xor(p0, 1); p0 += __shfl_xor(p0, 2); p0 += __shfl_xor(p0, 4); p0 += __shfl_xor(p0, 8);
      p1 += __shfl_xor(p1, 1); p1 += __shfl_xor(p1, 2); p1 += __shfl_xor(p1, 4); p1 += __shfl_xor(p1, 8);
      p2 += __shfl_xor(p2, 1); p2 += __shfl_xor(p2, 2); p2 += __shfl_xor(p2, 4); p2 += __shfl_xor(p2, 8);
      p3 += __shfl_xor(p3, 1); p3 += __shfl_xor(p3, 2); p3 += __shfl_xor(p3, 4); p3 += __shfl_xor(p3, 8);
      const float z0 = fmaxf(p0 + b1[d * 4 + 0], 0.f);
      const float z1 = fmaxf(p1 + b1[d * 4 + 1], 0.f);
      const float z2 = fmaxf(p2 + b1[d * 4 + 2], 0.f);
      const float z3 = fmaxf(p3 + b1[d * 4 + 3], 0.f);

      // h += z @ W2 + b2
      const float* w2p = W2 + d * 4 * HID + kbase;
      const float* b2p = b2 + d * HID + kbase;
      #pragma unroll
      for (int i = 0; i < 32; ++i) {
        hv[i] += z0 * w2p[i] + z1 * w2p[HID + i] + z2 * w2p[2 * HID + i]
               + z3 * w2p[3 * HID + i] + b2p[i];
      }
    }

    // write h back as bf16 into xbuf for the head GEMM
    #pragma unroll
    for (int i = 0; i < 32; i += 4) {
      v4bf b; b[0] = f2bf(hv[i]); b[1] = f2bf(hv[i+1]);
              b[2] = f2bf(hv[i+2]); b[3] = f2bf(hv[i+3]);
      *(v4bf*)&xbuf[myrow * XSTR + kbase + i] = b;
    }
  }
  __syncthreads();

  // ---- Phase 3: out = h @ W_head + b_head ----
  #pragma unroll
  for (int mt = 0; mt < 2; ++mt)
    #pragma unroll
    for (int nt = 0; nt < 4; ++nt)
      acc[mt][nt] = v8f{0.f,0.f,0.f,0.f,0.f,0.f,0.f,0.f};
  gemm_tile(xbuf, WheadT, acc, w, lane);
  #pragma unroll
  for (int mt = 0; mt < 2; ++mt) {
    #pragma unroll
    for (int nt = 0; nt < 4; ++nt) {
      const int col = (w << 6) + (nt << 4) + (lane & 15);
      const float bc = b_head[col];
      #pragma unroll
      for (int r = 0; r < 8; ++r) {
        const int row = mt * 16 + r + ((lane >> 4) << 3);
        __builtin_nontemporal_store(acc[mt][nt][r] + bc,
                                    &out[(row0 + row) * HID + col]);
      }
    }
  }
}

extern "C" void kernel_launch(void* const* d_in, const int* in_sizes, int n_in,
                              void* d_out, int out_size, void* d_ws, size_t ws_size,
                              hipStream_t stream) {
  const float* x      = (const float*)d_in[0];
  const float* W_in   = (const float*)d_in[1];
  const float* b_in   = (const float*)d_in[2];
  const float* ln_g   = (const float*)d_in[3];
  const float* ln_b   = (const float*)d_in[4];
  const float* W1     = (const float*)d_in[5];
  const float* b1     = (const float*)d_in[6];
  const float* W2     = (const float*)d_in[7];
  const float* b2     = (const float*)d_in[8];
  const float* W_head = (const float*)d_in[9];
  const float* b_head = (const float*)d_in[10];
  float* out = (float*)d_out;

  const int B = in_sizes[0] / HID;            // 131072

  __bf16* WinT   = (__bf16*)d_ws;                         // 512*512 bf16
  __bf16* WheadT = (__bf16*)((char*)d_ws + HID * HID * 2);

  prep_weights<<<(HID * HID) / 256, 256, 0, stream>>>(W_in, W_head, WinT, WheadT);
  fused_residual_mlp<<<B / MT, 256, 0, stream>>>(
      x, WinT, b_in, ln_g, ln_b, W1, b1, W2, b2, WheadT, b_head, out);
}
